// LSTM_10694468567791
// MI455X (gfx1250) — matrix operand order repaired and checked by reading, not compiled, not run on hardware
//
#include <hip/hip_runtime.h>

// ---------------------------------------------------------------------------
// LSTM (3 layers, S=8192, IN=H=1024, batch=1) for MI455X / gfx1250 (wave32).
// Phase A: bf16 WMMA GEMM xg = X @ Wih^T + b, block-tiled 64x128, LDS double
//          buffered via GLOBAL_LOAD_ASYNC_TO_LDS_B128 (ASYNCcnt, inline asm).
// Phase B: persistent recurrence kernel, Whh slice pinned in LDS (256KB/WGP).
// Phase C: final projection dot product.
// ---------------------------------------------------------------------------

#define SEQ   8192
#define HID   1024
#define GATES 4096            // 4*HID
#define REC_NB 64             // persistent blocks in recurrence
#define JPB   16              // hidden units per recurrence block (HID/REC_NB)

#define BM 64                 // GEMM block tile M
#define BN 128                // GEMM block tile N
#define BK 32                 // GEMM K step (one bf16 WMMA depth)

typedef __attribute__((ext_vector_type(16))) __bf16 v16bf;
typedef __attribute__((ext_vector_type(8)))  float  v8f;

// ---------------------------------------------------------------- conversion
__device__ __forceinline__ unsigned short f32_to_bf16_rne(float f) {
    unsigned u = __float_as_uint(f);
    u += 0x7fffu + ((u >> 16) & 1u);       // round-to-nearest-even
    return (unsigned short)(u >> 16);
}

__global__ void cvt_f32_bf16(const float* __restrict__ in,
                             unsigned short* __restrict__ out, int n) {
    int i = blockIdx.x * blockDim.x + threadIdx.x;
    if (i < n) out[i] = f32_to_bf16_rne(in[i]);
}

// ----------------------------------------------------------- async staging
// GLOBAL_LOAD_ASYNC_TO_LDS_B128 (ISA 15.18.3 op 98): copies 16 bytes from global memory to the LDS address in vdst,
// tracked with ASYNCcnt. LDS byte address = low 32 bits of generic pointer
// (ISA 10.2 aperture mapping).
__device__ __forceinline__ void stage_b128(const unsigned short* g,
                                           unsigned short* l) {
    unsigned lds_addr = (unsigned)(unsigned long long)(void*)l;
    asm volatile("global_load_async_to_lds_b128 %0, %1, off"
                 :: "v"(lds_addr), "v"(g)
                 : "memory");
}

__device__ __forceinline__ void wait_async_le3() {
#if defined(__has_builtin) && __has_builtin(__builtin_amdgcn_s_wait_asynccnt)
    __builtin_amdgcn_s_wait_asynccnt(3);
#else
    asm volatile("s_wait_asynccnt 0x3" ::: "memory");
#endif
}

__device__ __forceinline__ void wait_async_0() {
#if defined(__has_builtin) && __has_builtin(__builtin_amdgcn_s_wait_asynccnt)
    __builtin_amdgcn_s_wait_asynccnt(0);
#else
    asm volatile("s_wait_asynccnt 0x0" ::: "memory");
#endif
}

// ------------------------------------------------------------------ WMMA GEMM
// xg[M=SEQ, N=GATES] = Xb[M,K] * Wb[N,K]^T + (bih + bhh)[N]
// Block: 256 thr = 8 waves as 2x4; each wave computes 32x32 (2x2 WMMA tiles).
__global__ void gemm_bf16_wmma(const unsigned short* __restrict__ Xb,
                               const unsigned short* __restrict__ Wb,
                               const float* __restrict__ bih,
                               const float* __restrict__ bhh,
                               float* __restrict__ xg) {
    __shared__ unsigned short Asm[2][BM * BK];   // 2 x 4 KB
    __shared__ unsigned short Bsm[2][BN * BK];   // 2 x 8 KB

    const int K    = HID;
    const int tid  = threadIdx.x;
    const int lane = tid & 31;
    const int wv   = tid >> 5;          // wave 0..7
    const int wr   = wv >> 2;           // 0..1  wave-row
    const int wc   = wv & 3;            // 0..3  wave-col
    const int r    = lane & 15;
    const int half = lane >> 4;

    const int bm = (int)(blockIdx.x >> 5) * BM;   // 128 block rows
    const int bn = (int)(blockIdx.x & 31) * BN;   // 32 block cols

    // Per-thread staging: 1 A b128 + 2 B b128 per K step.
    const int arow = tid >> 2, aq = tid & 3;      // A tile: 64 rows x 4 b128
    const unsigned short* ag = Xb + (size_t)(bm + arow) * K + aq * 8;
    unsigned short* al = &Asm[0][arow * BK + aq * 8];
    const int brow = tid >> 2, bq = tid & 3;      // B tile: 128 rows x 4 b128
    const unsigned short* bg0 = Wb + (size_t)(bn + brow) * K + bq * 8;
    const unsigned short* bg1 = Wb + (size_t)(bn + 64 + brow) * K + bq * 8;
    unsigned short* bl0 = &Bsm[0][brow * BK + bq * 8];
    unsigned short* bl1 = &Bsm[0][(64 + brow) * BK + bq * 8];

    v8f acc00 = {}, acc01 = {}, acc10 = {}, acc11 = {};

    const int nk = K / BK;                         // 32

    // prologue: stage k-step 0 into buffer 0
    stage_b128(ag, al);
    stage_b128(bg0, bl0);
    stage_b128(bg1, bl1);

    for (int ks = 0; ks < nk; ++ks) {
        const int buf = ks & 1;
        const bool pre = (ks + 1 < nk);
        if (pre) {
            const int k1 = (ks + 1) * BK;
            const int off = (buf ^ 1) ? 1 : 0;
            stage_b128(ag + k1, al + off * (BM * BK));
            stage_b128(bg0 + k1, bl0 + off * (BN * BK));
            stage_b128(bg1 + k1, bl1 + off * (BN * BK));
            wait_async_le3();        // previous buffer's 3 ops are done
        } else {
            wait_async_0();
        }
        __syncthreads();             // all lanes' tiles visible in LDS

        // fragments from LDS (ds_load_b128)
        union U { uint4 q[2]; v16bf v; };
        U A0, A1, B0, B1;
        const unsigned short* aP0 = &Asm[buf][(wr * 32 + r) * BK + half * 8];
        const unsigned short* aP1 = &Asm[buf][(wr * 32 + 16 + r) * BK + half * 8];
        const unsigned short* bP0 = &Bsm[buf][(wc * 32 + r) * BK + half * 8];
        const unsigned short* bP1 = &Bsm[buf][(wc * 32 + 16 + r) * BK + half * 8];
        A0.q[0] = *(const uint4*)(aP0);      A0.q[1] = *(const uint4*)(aP0 + 16);
        A1.q[0] = *(const uint4*)(aP1);      A1.q[1] = *(const uint4*)(aP1 + 16);
        B0.q[0] = *(const uint4*)(bP0);      B0.q[1] = *(const uint4*)(bP0 + 16);
        B1.q[0] = *(const uint4*)(bP1);      B1.q[1] = *(const uint4*)(bP1 + 16);

        acc00 = __builtin_amdgcn_wmma_f32_16x16x32_bf16(
                    false, A0.v, false, B0.v, (short)0, acc00, false, false);
        acc01 = __builtin_amdgcn_wmma_f32_16x16x32_bf16(
                    false, A0.v, false, B1.v, (short)0, acc01, false, false);
        acc10 = __builtin_amdgcn_wmma_f32_16x16x32_bf16(
                    false, A1.v, false, B0.v, (short)0, acc10, false, false);
        acc11 = __builtin_amdgcn_wmma_f32_16x16x32_bf16(
                    false, A1.v, false, B1.v, (short)0, acc11, false, false);

        __syncthreads();             // tile consumed; safe to restage
    }

    // epilogue: bias + store (ISA C/D layout: row = i*16 + half*8 + v, col = r)
    const int col0 = bn + wc * 32 + r;
    const int col1 = col0 + 16;
    const float bias0 = bih[col0] + bhh[col0];
    const float bias1 = bih[col1] + bhh[col1];
#pragma unroll
    for (int v = 0; v < 8; ++v) {
        int row0 = bm + wr * 32 + half * 8 + v;
        int row1 = row0 + 16;
        xg[(size_t)row0 * GATES + col0] = acc00[v] + bias0;
        xg[(size_t)row0 * GATES + col1] = acc01[v] + bias1;
        xg[(size_t)row1 * GATES + col0] = acc10[v] + bias0;
        xg[(size_t)row1 * GATES + col1] = acc11[v] + bias1;
    }
}

// ------------------------------------------------------------- grid barrier
__device__ __forceinline__ void grid_barrier(int* cnt, volatile int* gen,
                                             int nblocks) {
    __syncthreads();
    if (threadIdx.x == 0) {
        __threadfence();
        int g = *gen;
        if (atomicAdd(cnt, 1) == nblocks - 1) {
            *cnt = 0;
            __threadfence();
            atomicAdd((int*)gen, 1);
        } else {
            while (*gen == g) { __builtin_amdgcn_s_sleep(1); }
        }
        __threadfence();
    }
    __syncthreads();
}

__global__ void init_rec(float* hbuf, int* sync) {
    int i = blockIdx.x * blockDim.x + threadIdx.x;
    if (i < 2 * HID) hbuf[i] = 0.0f;
    if (i == 0) { sync[0] = 0; sync[1] = 0; }
}

// -------------------------------------------------------------- recurrence
// Block b owns hidden units j in [b*16, b*16+16): 64 Whh rows staged in LDS
// (256 KB). Ping-pong h buffers in global; cell state lives in LDS.
__global__ void lstm_rec(const float* __restrict__ xg,
                         const float* __restrict__ Whh,
                         float* __restrict__ hseq,
                         float* __restrict__ hbuf,
                         int* __restrict__ sync) {
    extern __shared__ float smem[];
    float* wlds = smem;                     // 64*1024 floats (256 KB)
    float* red  = wlds + 64 * HID;          // 4*16*16 = 1024
    float* gv   = red + 1024;               // 64 gate values
    float* cst  = gv + 64;                  // 16 cell states

    const int tid = threadIdx.x;
    const int jb  = blockIdx.x * JPB;

    // Stage this block's 64 Whh rows into LDS (rows ordered [gate][j_local]).
    for (int fid = tid; fid < (64 * HID) / 4; fid += 256) {
        int row_local = fid >> 8;           // 256 float4 per row
        int k4 = fid & 255;
        int g  = row_local >> 4;
        int jl = row_local & 15;
        ((float4*)wlds)[fid] =
            ((const float4*)(Whh + (size_t)(g * HID + jb + jl) * HID))[k4];
    }
    if (tid < JPB) cst[tid] = 0.0f;
    __syncthreads();

    const int jl  = tid & 15;               // hidden unit within block
    const int sub = tid >> 4;               // K-slice 0..15 (64 k each)
    const float* w0 = wlds + (0 * 16 + jl) * HID;
    const float* w1 = wlds + (1 * 16 + jl) * HID;
    const float* w2 = wlds + (2 * 16 + jl) * HID;
    const float* w3 = wlds + (3 * 16 + jl) * HID;

    for (int t = 0; t < SEQ; ++t) {
        const float* hr = hbuf + (t & 1) * HID;       // h_{t-1}
        float a0 = 0.f, a1 = 0.f, a2 = 0.f, a3 = 0.f;
        int kbeg = sub * 64;
#pragma unroll 4
        for (int k = kbeg; k < kbeg + 64; k += 4) {
            float4 h4 = *(const float4*)(hr + k);
            float4 x0 = *(const float4*)(w0 + k);
            float4 x1 = *(const float4*)(w1 + k);
            float4 x2 = *(const float4*)(w2 + k);
            float4 x3 = *(const float4*)(w3 + k);
            a0 += x0.x*h4.x + x0.y*h4.y + x0.z*h4.z + x0.w*h4.w;
            a1 += x1.x*h4.x + x1.y*h4.y + x1.z*h4.z + x1.w*h4.w;
            a2 += x2.x*h4.x + x2.y*h4.y + x2.z*h4.z + x2.w*h4.w;
            a3 += x3.x*h4.x + x3.y*h4.y + x3.z*h4.z + x3.w*h4.w;
        }
        red[(0 * 16 + jl) * 16 + sub] = a0;
        red[(1 * 16 + jl) * 16 + sub] = a1;
        red[(2 * 16 + jl) * 16 + sub] = a2;
        red[(3 * 16 + jl) * 16 + sub] = a3;
        __syncthreads();

        if (tid < 64) {                     // tid == g*16 + j
            int g = tid >> 4, j = tid & 15;
            const float* rr = red + tid * 16;
            float s = 0.f;
#pragma unroll
            for (int u = 0; u < 16; ++u) s += rr[u];
            gv[tid] = s + xg[(size_t)t * GATES + g * HID + jb + j];
        }
        __syncthreads();

        if (tid < JPB) {
            float gi = gv[tid];
            float gf = gv[16 + tid];
            float gg = gv[32 + tid];
            float go = gv[48 + tid];
            float i = 1.f / (1.f + __expf(-gi));
            float f = 1.f / (1.f + __expf(-gf));
            float o = 1.f / (1.f + __expf(-go));
            float cn = f * cst[tid] + i * tanhf(gg);
            float hn = o * tanhf(cn);
            cst[tid] = cn;
            hbuf[((t & 1) ^ 1) * HID + jb + tid] = hn;   // h_t
            hseq[(size_t)t * HID + jb + tid]     = hn;
        }
        __threadfence();
        grid_barrier(sync, (volatile int*)(sync + 1), REC_NB);
    }
}

// ------------------------------------------------------------- projection
__global__ void proj_out(const float* __restrict__ hlast,
                         const float* __restrict__ Wout,
                         const float* __restrict__ bout,
                         float* __restrict__ out) {
    __shared__ float s[256];
    float a = 0.f;
    for (int k = threadIdx.x; k < HID; k += 256) a += hlast[k] * Wout[k];
    s[threadIdx.x] = a;
    __syncthreads();
    for (int w = 128; w > 0; w >>= 1) {
        if (threadIdx.x < w) s[threadIdx.x] += s[threadIdx.x + w];
        __syncthreads();
    }
    if (threadIdx.x == 0) out[0] = s[0] + bout[0];
}

// ---------------------------------------------------------------- launcher
extern "C" void kernel_launch(void* const* d_in, const int* in_sizes, int n_in,
                              void* d_out, int out_size, void* d_ws, size_t ws_size,
                              hipStream_t stream) {
    const float* x_in  = (const float*)d_in[0];   // [S, IN]
    const float* W_ih  = (const float*)d_in[1];   // [L, 4H, IN]
    const float* W_hh  = (const float*)d_in[2];   // [L, 4H, H]
    const float* b_ih  = (const float*)d_in[3];   // [L, 4H]
    const float* b_hh  = (const float*)d_in[4];   // [L, 4H]
    const float* W_out = (const float*)d_in[5];   // [1, H]
    const float* b_out = (const float*)d_in[6];   // [1]
    float* out = (float*)d_out;

    char* ws = (char*)d_ws;
    float* xg   = (float*)ws;           ws += (size_t)SEQ * GATES * 4;   // 128 MB
    float* hseq = (float*)ws;           ws += (size_t)SEQ * HID * 4;     //  32 MB
    unsigned short* xb = (unsigned short*)ws; ws += (size_t)SEQ * HID * 2;   // 16 MB
    unsigned short* wb = (unsigned short*)ws; ws += (size_t)GATES * HID * 2; //  8 MB
    float* hbuf = (float*)ws;           ws += 2 * HID * 4;
    int*   sync = (int*)ws;             ws += 256;

    const size_t lds_bytes = (64 * HID + 1024 + 64 + 16) * sizeof(float);
    const int gemm_blocks  = (SEQ / BM) * (GATES / BN);       // 128*32 = 4096

    const float* cur = x_in;
    for (int l = 0; l < 3; ++l) {
        cvt_f32_bf16<<<(SEQ * HID + 255) / 256, 256, 0, stream>>>(cur, xb, SEQ * HID);
        cvt_f32_bf16<<<(GATES * HID + 255) / 256, 256, 0, stream>>>(
            W_ih + (size_t)l * GATES * HID, wb, GATES * HID);
        gemm_bf16_wmma<<<gemm_blocks, 256, 0, stream>>>(
            xb, wb, b_ih + (size_t)l * GATES, b_hh + (size_t)l * GATES, xg);
        init_rec<<<8, 256, 0, stream>>>(hbuf, sync);
        lstm_rec<<<REC_NB, 256, lds_bytes, stream>>>(
            xg, W_hh + (size_t)l * GATES * HID, hseq, hbuf, sync);
        cur = hseq;
    }
    proj_out<<<1, 256, 0, stream>>>(hseq + (size_t)(SEQ - 1) * HID, W_out, b_out, out);
}
